// RNNRBM_NEW_64613488001652
// MI455X (gfx1250) — matrix operand order repaired and checked by reading, not compiled, not run on hardware
//
#include <hip/hip_runtime.h>
#include <hip/hip_bf16.h>
#include <math.h>

// ---------------------------------------------------------------------------
// RNN-RBM training step for MI455X (gfx1250, wave32, WMMA).
// Gibbs chains are independent per time step -> batched WMMA GEMMs over all
// 511 chains; only the small u-recurrence is serial (1 WGP, u in LDS).
// GEMM kernel templated on <AT,BT,MODE>:
//  - contiguous-K operands: global_load_b128 -> v_cvt_pk_f16_f32 -> wmma
//  - K-strided B operands : async global->LDS staging
//    (global_load_async_to_lds_b128 / s_wait_asynccnt) + swizzled LDS reads
//  - out-of-range rows/cols are CLAMPED, not zero-filled: A row m only feeds
//    D row m (and B col n only D col n), which the epilogue never stores.
// ---------------------------------------------------------------------------

typedef __attribute__((ext_vector_type(16))) _Float16 v16h;
typedef __attribute__((ext_vector_type(8)))  float    v8f;

#define VD 256
#define HD 512
#define RD 512
#define TT 512
#define TM 511   /* T-1 */
#define TMID 510 /* T-2 */

// ---- stateless RNG: splitmix64-style hash -> uniform [0,1) ---------------
__device__ inline float rng_uniform(unsigned long long seed, unsigned idx) {
  unsigned long long z = seed + 0x9E3779B97F4A7C15ull * (unsigned long long)(idx + 1u);
  z = (z ^ (z >> 30)) * 0xBF58476D1CE4E5B9ull;
  z = (z ^ (z >> 27)) * 0x94D049BB133111EBull;
  z ^= z >> 31;
  return (float)(unsigned)(z >> 40) * (1.0f / 16777216.0f);
}

__device__ inline float softplusf(float x) {
  return (x > 20.f) ? x : log1pf(expf(x));
}

// ===========================================================================
// WMMA fragment loaders (wave32; A and B share contiguous k order K = 16*(L>>4)+e).
// A 16x32 f16: lane L row m0+(L&15); halves 0..7  -> K = (L<16?0:16)+e,
//              halves 8..15 -> K = (L<16?8:24)+(e-8).
// B 32x16 f16: lane L col n0+(L&15); halves e: K = (L<16?0:16)+e.
// AT==0: A row-major (m,k)@A[m*lda+k]; AT==1: K-major (m,k)@A[k*lda+m].
// BT==0: B row-major KxN (k,n)@B[k*ldb+n]; BT==1: NxK (k,n)@B[n*ldb+k].
// ===========================================================================

template <int AT>
__device__ inline v16h load_a_fast(const float* __restrict__ A, int lda,
                                   int m0, int k0, int M, int lane) {
  int m = m0 + (lane & 15);
  m = (m < M) ? m : (M - 1);  // clamp: row m only feeds unstored D rows
  int hi  = lane >> 4;
  int kb0 = k0 + (hi ? 16 : 0);
  int kb1 = k0 + (hi ? 24 : 8);
  v16h af;
  if (AT == 0) {
    const float* p = A + (size_t)m * lda;
    float4 x0 = *(const float4*)(p + kb0);
    float4 x1 = *(const float4*)(p + kb0 + 4);
    float4 x2 = *(const float4*)(p + kb1);
    float4 x3 = *(const float4*)(p + kb1 + 4);
    af[0] = (_Float16)x0.x;  af[1] = (_Float16)x0.y;
    af[2] = (_Float16)x0.z;  af[3] = (_Float16)x0.w;
    af[4] = (_Float16)x1.x;  af[5] = (_Float16)x1.y;
    af[6] = (_Float16)x1.z;  af[7] = (_Float16)x1.w;
    af[8] = (_Float16)x2.x;  af[9] = (_Float16)x2.y;
    af[10] = (_Float16)x2.z; af[11] = (_Float16)x2.w;
    af[12] = (_Float16)x3.x; af[13] = (_Float16)x3.y;
    af[14] = (_Float16)x3.z; af[15] = (_Float16)x3.w;
  } else {
#pragma unroll
    for (int e = 0; e < 8; ++e)
      af[e] = (_Float16)A[(size_t)(kb0 + e) * lda + m];
#pragma unroll
    for (int e = 0; e < 8; ++e)
      af[e + 8] = (_Float16)A[(size_t)(kb1 + e) * lda + m];
  }
  return af;
}

template <int AT>
__device__ inline v16h load_a_safe(const float* __restrict__ A, int lda,
                                   int m0, int k0, int M, int K, int lane) {
  int m = m0 + (lane & 15);
  m = (m < M) ? m : (M - 1);
  int hi  = lane >> 4;
  int kb0 = k0 + (hi ? 16 : 0);
  int kb1 = k0 + (hi ? 24 : 8);
  v16h af;
#pragma unroll
  for (int e = 0; e < 8; ++e) {
    int k = kb0 + e;
    float x = (k < K) ? (AT ? A[(size_t)k * lda + m] : A[(size_t)m * lda + k])
                      : 0.f;
    af[e] = (_Float16)x;
  }
#pragma unroll
  for (int e = 0; e < 8; ++e) {
    int k = kb1 + e;
    float x = (k < K) ? (AT ? A[(size_t)k * lda + m] : A[(size_t)m * lda + k])
                      : 0.f;
    af[e + 8] = (_Float16)x;
  }
  return af;
}

// contiguous-K B fragment (BT==1 only in fast path)
__device__ inline v16h load_b_fast_nk(const float* __restrict__ B, int ldb,
                                      int n0, int k0, int N, int lane) {
  int n = n0 + (lane & 15);
  n = (n < N) ? n : (N - 1);  // clamp: col n only feeds unstored D cols
  int kb = k0 + ((lane >> 4) ? 16 : 0);
  const float* p = B + (size_t)n * ldb + kb;
  float4 x0 = *(const float4*)(p + 0);
  float4 x1 = *(const float4*)(p + 4);
  float4 x2 = *(const float4*)(p + 8);
  float4 x3 = *(const float4*)(p + 12);
  v16h bf;
  bf[0] = (_Float16)x0.x;  bf[1] = (_Float16)x0.y;
  bf[2] = (_Float16)x0.z;  bf[3] = (_Float16)x0.w;
  bf[4] = (_Float16)x1.x;  bf[5] = (_Float16)x1.y;
  bf[6] = (_Float16)x1.z;  bf[7] = (_Float16)x1.w;
  bf[8] = (_Float16)x2.x;  bf[9] = (_Float16)x2.y;
  bf[10] = (_Float16)x2.z; bf[11] = (_Float16)x2.w;
  bf[12] = (_Float16)x3.x; bf[13] = (_Float16)x3.y;
  bf[14] = (_Float16)x3.z; bf[15] = (_Float16)x3.w;
  return bf;
}

template <int BT>
__device__ inline v16h load_b_safe(const float* __restrict__ B, int ldb,
                                   int n0, int k0, int N, int K, int lane) {
  int n = n0 + (lane & 15);
  n = (n < N) ? n : (N - 1);
  int kb = k0 + ((lane >> 4) ? 16 : 0);
  v16h bf;
#pragma unroll
  for (int e = 0; e < 16; ++e) {
    int k = kb + e;
    float x = (k < K) ? (BT ? B[(size_t)n * ldb + k] : B[(size_t)k * ldb + n])
                      : 0.f;
    bf[e] = (_Float16)x;
  }
  return bf;
}

// ---- flexible WMMA GEMM: C(MxN) = epilogue( sum_k A(m,k)*B(k,n) ) ---------
// MODE 0: C = alpha*acc + beta*C      (beta==0 -> no read)
// MODE 1: C = acc + aux1[n]           (bias over N)
// MODE 2: C = Bernoulli( sigmoid(acc + aux1[m*auxld+n]) )   (Gibbs sample)
// MODE 3: C = acc + aux1[m]*aux2[n]   (rank-1 add)
template <int AT, int BT, int MODE>
__global__ __launch_bounds__(128) void wmma_gemm_kernel(
    const float* __restrict__ A, int lda, const float* __restrict__ B, int ldb,
    float* __restrict__ C, int ldc, int M, int N, int K,
    float alpha, float beta,
    const float* __restrict__ aux1, const float* __restrict__ aux2,
    int auxld, unsigned long long seed) {
  int lane = threadIdx.x & 31;
  int wave = threadIdx.x >> 5;
  int m0 = blockIdx.y * 32 + ((wave >> 1) << 4);
  int n0 = blockIdx.x * 32 + ((wave & 1) << 4);

  v8f acc0 = {}, acc1 = {};
  const int kfull = K & ~31;
  int k0 = 0;

  if constexpr (BT == 0) {
    // ---- K-strided B: stage 32x32 B slab into LDS via async copy ----------
    // layout bsh[k][n] with column swizzle n ^ (((k>>4)&1)<<4): 16B-aligned
    // rows (128B stride) and conflict-free half-wave reads.
    __shared__ float bsh[32][32];
    const int tid  = threadIdx.x;
    const int nblk = blockIdx.x * 32;
    const int nl   = ((wave & 1) << 4) + (lane & 15);
    const int kb   = (lane >> 4) ? 16 : 0;
    const int colr = nl ^ (kb ? 16 : 0);
    for (; k0 < kfull; k0 += 32) {
      __syncthreads();  // previous slab fully consumed
#pragma unroll
      for (int i = 0; i < 2; ++i) {
        int f   = tid * 2 + i;          // 0..255 float4 tiles
        int row = f >> 3;               // k-local 0..31
        int c4  = (f & 7) << 2;         // n-local {0,4,...,28}
        int col = c4 ^ (((row >> 4) & 1) << 4);
        const float* gp = B + (size_t)(k0 + row) * ldb + (nblk + c4);
        unsigned lp = (unsigned)(size_t)(void*)&bsh[row][col];
        asm volatile("global_load_async_to_lds_b128 %0, %1, off"
                     :: "v"(lp), "v"(gp) : "memory");
      }
      asm volatile("s_wait_asynccnt 0x0" ::: "memory");
      __syncthreads();  // all waves' async data visible
      v16h a0 = load_a_fast<AT>(A, lda, m0, k0, M, lane);
      v16h b0;
#pragma unroll
      for (int e = 0; e < 16; ++e) b0[e] = (_Float16)bsh[kb + e][colr];
      acc0 = __builtin_amdgcn_wmma_f32_16x16x32_f16(false, a0, false, b0,
                                                    (short)0, acc0, false,
                                                    false);
    }
  } else {
    // ---- contiguous-K B: vector loads, 2 K-slabs/iter, 2 accumulators -----
    for (; k0 + 64 <= kfull; k0 += 64) {
      if (k0 + 128 <= kfull) {  // speculative prefetch (global_prefetch_b8)
        const float* pa = AT ? (A + (size_t)(k0 + 64) * lda + m0)
                             : (A + (size_t)m0 * lda + (k0 + 64));
        const float* pb = B + (size_t)n0 * ldb + (k0 + 64);
        __builtin_prefetch(pa, 0, 1);
        __builtin_prefetch(pb, 0, 1);
      }
      v16h a0 = load_a_fast<AT>(A, lda, m0, k0, M, lane);
      v16h b0 = load_b_fast_nk(B, ldb, n0, k0, N, lane);
      acc0 = __builtin_amdgcn_wmma_f32_16x16x32_f16(false, a0, false, b0,
                                                    (short)0, acc0, false,
                                                    false);
      v16h a1 = load_a_fast<AT>(A, lda, m0, k0 + 32, M, lane);
      v16h b1 = load_b_fast_nk(B, ldb, n0, k0 + 32, N, lane);
      acc1 = __builtin_amdgcn_wmma_f32_16x16x32_f16(false, a1, false, b1,
                                                    (short)0, acc1, false,
                                                    false);
    }
    if (k0 < kfull) {
      v16h a0 = load_a_fast<AT>(A, lda, m0, k0, M, lane);
      v16h b0 = load_b_fast_nk(B, ldb, n0, k0, N, lane);
      acc0 = __builtin_amdgcn_wmma_f32_16x16x32_f16(false, a0, false, b0,
                                                    (short)0, acc0, false,
                                                    false);
      k0 += 32;
    }
  }

  // K tail (bounds-checked, zero-filled beyond K)
  if (k0 < K) {
    v16h a1 = load_a_safe<AT>(A, lda, m0, k0, M, K, lane);
    v16h b1 = load_b_safe<BT>(B, ldb, n0, k0, N, K, lane);
    acc1 = __builtin_amdgcn_wmma_f32_16x16x32_f16(false, a1, false, b1,
                                                  (short)0, acc1, false,
                                                  false);
  }

  int nn = n0 + (lane & 15);
#pragma unroll
  for (int r = 0; r < 8; ++r) {
    int mm = m0 + r + ((lane >> 4) << 3);
    if (mm < M && nn < N) {
      float v = acc0[r] + acc1[r];
      if (MODE == 0) {
        v = alpha * v + ((beta != 0.f) ? beta * C[(size_t)mm * ldc + nn] : 0.f);
      } else if (MODE == 1) {
        v = v + aux1[nn];
      } else if (MODE == 2) {
        float x = v + aux1[(size_t)mm * auxld + nn];
        float p = 1.f / (1.f + expf(-x));
        v = (p > rng_uniform(seed, (unsigned)(mm * N + nn))) ? 1.f : 0.f;
      } else {
        v = v + aux1[mm] * aux2[nn];
      }
      C[(size_t)mm * ldc + nn] = v;
    }
  }
}

// ---- serial u recurrence: AllUt[0]=AllUt[1]=u0, AllUt[t+1]=u_t ------------
__global__ __launch_bounds__(512) void u_recurrence_kernel(
    const float* __restrict__ Xvu, const float* __restrict__ wuu,
    const float* __restrict__ bu, const float* __restrict__ u0,
    float* __restrict__ AllUt) {
  __shared__ float u[RD];
  int tid = threadIdx.x;  // 512 threads, one per r
  float uv = u0[tid];
  u[tid] = uv;
  AllUt[tid] = uv;       // all_ut[0] = u0
  AllUt[RD + tid] = uv;  // all_ut[1] = u0 (scan stacks the pre-update carry)
  __syncthreads();
  for (int t = 1; t <= TMID; ++t) {  // u_1 .. u_510
    const float4* wr = (const float4*)(wuu + (size_t)tid * RD);
    float s = 0.f;
#pragma unroll 4
    for (int j = 0; j < RD / 4; ++j) {
      float4 wv = wr[j];
      s += wv.x * u[4 * j + 0] + wv.y * u[4 * j + 1] +
           wv.z * u[4 * j + 2] + wv.w * u[4 * j + 3];
    }
    float ut = tanhf(Xvu[(size_t)t * RD + tid] + s + bu[tid]);
    __syncthreads();
    u[tid] = ut;
    AllUt[(size_t)(t + 1) * RD + tid] = ut;
    __syncthreads();
  }
}

// ---- per-t statistics: tempA/tempB/dbht, free-energy diff, mse ------------
__global__ __launch_bounds__(256) void stats_kernel(
    const float* __restrict__ Anv, const float* __restrict__ Avt,
    const float* __restrict__ Bh, const float* __restrict__ bh,
    const float* __restrict__ bv, const float* __restrict__ NV,
    const float* __restrict__ visible, float* __restrict__ tA,
    float* __restrict__ tB, float* __restrict__ dbht,
    float* __restrict__ dbvt, float* __restrict__ mse_out) {
  int t = blockIdx.x;  // 0..510
  int j = threadIdx.x; // 256 threads
  float fe = 0.f, mse = 0.f;
#pragma unroll
  for (int hh = 0; hh < 2; ++hh) {
    int h = j + hh * 256;
    float anv = Anv[(size_t)t * HD + h];
    float avt = Avt[(size_t)t * HD + h];
    float bht = Bh[(size_t)t * HD + h];
    float a = 1.f / (1.f + expf(-(anv - bht)));
    float b = 1.f / (1.f + expf(-(avt - bht)));
    tA[(size_t)t * HD + h] = a;
    tB[(size_t)t * HD + h] = b;
    dbht[(size_t)t * HD + h] = a - b;
    fe += softplusf(avt + bh[h]) - softplusf(anv + bh[h]);
  }
  {
    int v = j;
    float nv = NV[(size_t)t * VD + v];
    float vt = visible[(size_t)(t + 1) * VD + v];
    fe += (vt - nv) * bv[v];
    mse = fabsf(vt - nv);
  }
  __shared__ float s1[256], s2[256];
  s1[j] = fe;
  s2[j] = mse;
  __syncthreads();
  for (int off = 128; off > 0; off >>= 1) {
    if (j < off) { s1[j] += s1[j + off]; s2[j] += s2[j + off]; }
    __syncthreads();
  }
  if (j == 0) {
    dbvt[t] = s1[0];
    mse_out[t] = s2[0] * (1.f / (float)VD);
  }
}

// ---- column sum over rows: out[c] = sum_r in[r*ld+c] ----------------------
__global__ void colreduce_kernel(const float* __restrict__ in, int rows,
                                 int cols, int ld, float* __restrict__ out) {
  int c = blockIdx.x * blockDim.x + threadIdx.x;
  if (c >= cols) return;
  float s = 0.f;
  for (int r = 0; r < rows; ++r) s += in[(size_t)r * ld + c];
  out[c] = s;
}

// ---- dbv: sum dbvt[0..510], broadcast scalar to 256-wide output -----------
__global__ __launch_bounds__(512) void dbv_kernel(const float* __restrict__ dbvt,
                                                  float* __restrict__ out) {
  __shared__ float sd[512];
  int j = threadIdx.x;
  sd[j] = (j < TM) ? dbvt[j] : 0.f;
  __syncthreads();
  for (int off = 256; off > 0; off >>= 1) {
    if (j < off) sd[j] += sd[j + off];
    __syncthreads();
  }
  float s = sd[0];
  if (j < VD) out[j] = s;
}

// ---- outer product: C[i*N+j] = a[i]*b[j] ----------------------------------
__global__ void outer_kernel(const float* __restrict__ a,
                             const float* __restrict__ b,
                             float* __restrict__ C, int M, int N) {
  int i = blockIdx.x * blockDim.x + threadIdx.x;
  if (i >= M * N) return;
  C[i] = a[i / N] * b[i % N];
}

// ---- temp = du * u_mid * (1 - u_mid), u_mid[t'] = AllUt[1+t'] -------------
__global__ void dtanh_kernel(const float* __restrict__ du,
                             const float* __restrict__ AllUt,
                             float* __restrict__ temp) {
  int i = blockIdx.x * blockDim.x + threadIdx.x;
  if (i >= TMID * RD) return;
  int t = i / RD, r = i % RD;
  float m = AllUt[(size_t)(t + 1) * RD + r];
  temp[i] = du[i] * m * (1.f - m);
}

// ---- host-side launcher ---------------------------------------------------
#define GEMM_LAUNCH(AT, BT, MODE, A, lda, B, ldb, C, ldc, M, N, K, alpha,   \
                    beta, a1, a2, auxld, seed)                              \
  do {                                                                      \
    dim3 g_((unsigned)(((N) + 31) / 32), (unsigned)(((M) + 31) / 32));      \
    wmma_gemm_kernel<AT, BT, MODE><<<g_, dim3(128), 0, stream>>>(           \
        A, lda, B, ldb, C, ldc, M, N, K, alpha, beta, a1, a2, auxld, seed); \
  } while (0)

extern "C" void kernel_launch(void* const* d_in, const int* in_sizes, int n_in,
                              void* d_out, int out_size, void* d_ws,
                              size_t ws_size, hipStream_t stream) {
  (void)in_sizes; (void)n_in; (void)out_size; (void)ws_size;
  const float* visible = (const float*)d_in[0];  // (T, VD)
  const float* w       = (const float*)d_in[1];  // (HD, VD)
  const float* wuu     = (const float*)d_in[2];  // (RD, RD)
  const float* wuv     = (const float*)d_in[3];  // (VD, RD)
  const float* wuh     = (const float*)d_in[4];  // (HD, RD)
  const float* wvu     = (const float*)d_in[5];  // (RD, VD)
  const float* bv      = (const float*)d_in[6];
  const float* bh      = (const float*)d_in[7];
  const float* bu      = (const float*)d_in[8];
  const float* u0      = (const float*)d_in[9];
  float* out = (float*)d_out;

  // output layout (flat, return order)
  float* o_mse  = out;                      // 511
  float* o_dw   = o_mse + TM;               // 512*256
  float* o_dbv  = o_dw + HD * VD;           // 256
  float* o_dbh  = o_dbv + VD;               // 512
  float* o_dwuv = o_dbh + HD;               // 256*512
  float* o_dwuh = o_dwuv + VD * RD;         // 512*512
  float* o_dbu  = o_dwuh + HD * RD;         // 512
  float* o_dwvu = o_dbu + RD;               // 512*256
  float* o_dwuu = o_dwvu + RD * VD;         // 512*512

  // workspace layout
  float* W = (float*)d_ws;
  float* AllUt  = W; W += TT * RD;
  float* Xvu    = W; W += TT * RD;
  float* Bhm    = W; W += TT * HD;
  float* Bvm    = W; W += TT * VD;
  float* Vcur   = W; W += TT * VD;
  float* Hm     = W; W += TT * HD;
  float* Anv    = W; W += TT * HD;
  float* Avt    = W; W += TT * HD;
  float* tA     = W; W += TT * HD;
  float* tB     = W; W += TT * HD;
  float* dbht   = W; W += TT * HD;
  float* duB    = W; W += TT * RD;
  float* tmp    = W; W += TT * RD;
  float* dbvt   = W; W += TT;
  float* colsum = W; W += RD;

  // 1) Xvu = visible @ wvu^T   (rows 1..510 used by recurrence)
  GEMM_LAUNCH(0, 1, 0, visible, VD, wvu, VD, Xvu, RD, TM, RD, VD, 1.f, 0.f,
              nullptr, nullptr, 0, 0ull);

  // 2) serial u recurrence -> AllUt (all_ut of the reference)
  u_recurrence_kernel<<<1, 512, 0, stream>>>(Xvu, wuu, bu, u0, AllUt);

  // 3) Bh[t] = wuh@u_t + bh ; Bv[t] = wuv@u_t + bv  (carry at step t is u_t)
  GEMM_LAUNCH(0, 1, 1, AllUt + RD, RD, wuh, RD, Bhm, HD, TM, HD, RD, 1.f, 0.f,
              bh, nullptr, 0, 0ull);
  GEMM_LAUNCH(0, 1, 1, AllUt + RD, RD, wuv, RD, Bvm, VD, TM, VD, RD, 1.f, 0.f,
              bv, nullptr, 0, 0ull);

  // 4) 20 Gibbs steps, all 511 chains batched per GEMM
  for (int s = 0; s < 20; ++s) {
    const float* Vin = (s == 0) ? visible : Vcur;
    GEMM_LAUNCH(0, 1, 2, Vin, VD, w, VD, Hm, HD, TM, HD, VD, 1.f, 0.f, Bhm,
                nullptr, HD, 0xC0FFEEull + 2ull * s);
    GEMM_LAUNCH(0, 0, 2, Hm, HD, w, VD, Vcur, VD, TM, VD, HD, 1.f, 0.f, Bvm,
                nullptr, VD, 0xC0FFEEull + 2ull * s + 1ull);
  }

  // 5) raw products for tempA/tempB and free energy
  GEMM_LAUNCH(0, 1, 0, Vcur, VD, w, VD, Anv, HD, TM, HD, VD, 1.f, 0.f,
              nullptr, nullptr, 0, 0ull);
  GEMM_LAUNCH(0, 1, 0, visible + VD, VD, w, VD, Avt, HD, TM, HD, VD, 1.f, 0.f,
              nullptr, nullptr, 0, 0ull);

  stats_kernel<<<TM, 256, 0, stream>>>(Anv, Avt, Bhm, bh, bv, Vcur, visible,
                                       tA, tB, dbht, dbvt, o_mse);

  // 6) dw = tempA^T @ NV  -  tempB^T @ V  (K = t, K-major operands)
  GEMM_LAUNCH(1, 0, 0, tA, HD, Vcur, VD, o_dw, VD, HD, VD, TM, 1.f, 0.f,
              nullptr, nullptr, 0, 0ull);
  GEMM_LAUNCH(1, 0, 0, tB, HD, visible + VD, VD, o_dw, VD, HD, VD, TM, -1.f,
              1.f, nullptr, nullptr, 0, 0ull);

  // 7) dbh, dbv, outer products with all_ut[T-1] = AllUt[511]
  colreduce_kernel<<<(HD + 255) / 256, 256, 0, stream>>>(dbht, TM, HD, HD,
                                                         o_dbh);
  dbv_kernel<<<1, 512, 0, stream>>>(dbvt, o_dbv);
  outer_kernel<<<(HD * RD + 255) / 256, 256, 0, stream>>>(
      o_dbh, AllUt + (size_t)(TT - 1) * RD, o_dwuh, HD, RD);
  outer_kernel<<<(VD * RD + 255) / 256, 256, 0, stream>>>(
      o_dbv, AllUt + (size_t)(TT - 1) * RD, o_dwuv, VD, RD);

  // 8) du = dbht[1:] @ wuh + dbvt[1:] * colsum(wuv)   (rank-1 folded in)
  colreduce_kernel<<<(RD + 255) / 256, 256, 0, stream>>>(wuv, VD, RD, RD,
                                                         colsum);
  GEMM_LAUNCH(0, 0, 3, dbht + HD, HD, wuh, RD, duB, RD, TMID, RD, HD, 1.f,
              0.f, dbvt + 1, colsum, 0, 0ull);

  // 9) temp = du * u_mid * (1-u_mid); dbu; dwuu; dwvu
  dtanh_kernel<<<(TMID * RD + 255) / 256, 256, 0, stream>>>(duB, AllUt, tmp);
  colreduce_kernel<<<(RD + 255) / 256, 256, 0, stream>>>(tmp, TMID, RD, RD,
                                                         o_dbu);
  GEMM_LAUNCH(1, 0, 0, tmp, RD, AllUt, RD, o_dwuu, RD, RD, RD, TMID, 1.f, 0.f,
              nullptr, nullptr, 0, 0ull);
  GEMM_LAUNCH(1, 0, 0, tmp, RD, visible + VD, VD, o_dwvu, VD, RD, VD, TMID,
              1.f, 0.f, nullptr, nullptr, 0, 0ull);
}
